// LinearAttention_65996467470824
// MI455X (gfx1250) — matrix operand order repaired and checked by reading, not compiled, not run on hardware
//
#include <hip/hip_runtime.h>

// Linear attention (softmax-feature / softmax-seq variant) for MI455X gfx1250.
// GEMMs: double-buffered LDS staged via global_load_async_to_lds_b128
// (ASYNCcnt) feeding v_wmma_f32_16x16x32_bf16 with fp32 accumulation.

#define NSEQ   4096
#define QKV3   3072
#define HEADS  16
#define DHEAD  64
#define HIDDEN 1024
#define ATT_SCALE 0.125f   // 64^-0.5

#define LDT    40          // LDS row stride in bf16 elements (80B)
#define BUFSZ  (128 * LDT * 2)  // bytes per stage buffer per matrix

typedef __attribute__((ext_vector_type(16))) __bf16 v16bf;
typedef __attribute__((ext_vector_type(8)))  __bf16 v8bf;
typedef __attribute__((ext_vector_type(8)))  float  v8f;

__device__ __forceinline__ unsigned short f2bf(float f) {
  union { float f; unsigned int u; } v; v.f = f;
  unsigned int u = v.u;
  u += 0x7FFFu + ((u >> 16) & 1u);          // round-to-nearest-even
  return (unsigned short)(u >> 16);
}

__device__ __forceinline__ void async_b128(unsigned lds_off, const void* gaddr) {
  asm volatile("global_load_async_to_lds_b128 %0, %1, off"
               :: "v"(lds_off), "v"(gaddr) : "memory");
}
__device__ __forceinline__ void async_b128_o16(unsigned lds_off, const void* gaddr) {
  asm volatile("global_load_async_to_lds_b128 %0, %1, off offset:16"
               :: "v"(lds_off), "v"(gaddr) : "memory");
}
__device__ __forceinline__ void wait_async0() {
#if __has_builtin(__builtin_amdgcn_s_wait_asynccnt)
  __builtin_amdgcn_s_wait_asynccnt(0);
#else
  asm volatile("s_wait_asynccnt 0x0" ::: "memory");
#endif
}

// ---------------------------------------------------------------- converts
__global__ __launch_bounds__(256) void cvt_bf16_kernel(const float* __restrict__ in,
                                                       unsigned short* __restrict__ out,
                                                       int n) {
  int i = blockIdx.x * 256 + threadIdx.x;
  if (i < n) out[i] = f2bf(in[i]);
}

// w[K][Nc] row-major fp32  ->  outT[Nc][K] bf16 (i.e. column-major B)
__global__ __launch_bounds__(256) void cvt_bf16_T_kernel(const float* __restrict__ w,
                                                         unsigned short* __restrict__ outT,
                                                         int K, int Nc) {
  int i = blockIdx.x * 256 + threadIdx.x;
  if (i < K * Nc) {
    int k = i / Nc, c = i % Nc;
    outT[(size_t)c * K + k] = f2bf(w[i]);
  }
}

// ---------------------------------------------------------------- GEMM
// C[M,Nn] f32 = A[M,K] bf16 row-major  x  Bt[Nn,K] bf16 (B column-major)
// block tile 128x128, 8 waves, wave tile 64(M)x32(N).
// K staged 32/iter into double-buffered LDS via async copies.
__global__ __launch_bounds__(256) void gemm_bf16_kernel(const unsigned short* __restrict__ A,
                                                        const unsigned short* __restrict__ Bt,
                                                        float* __restrict__ C,
                                                        const float* __restrict__ bias,
                                                        int M, int Nn, int K) {
  const int tid  = threadIdx.x;
  const int lane = tid & 31, wave = tid >> 5;
  const int wm = wave >> 2, wn = wave & 3;      // wave tile: rows wm*64, cols wn*32
  const int m = lane & 15, hi = lane >> 4;
  const int rowB0 = blockIdx.y * 128;
  const int colB0 = blockIdx.x * 128;

  __shared__ __align__(32) unsigned short As[2][128 * LDT];
  __shared__ __align__(32) unsigned short Bs[2][128 * LDT];

  // staging map: 2 threads per row, each covers 16 elems (two 16B chunks)
  const int srow  = tid >> 1;
  const int shalf = (tid & 1) * 16;             // element offset in 32-elem slab
  const unsigned ldsA0 = (unsigned)(size_t)(&As[0][0]) + (unsigned)(srow * LDT + shalf) * 2u;
  const unsigned ldsB0 = (unsigned)(size_t)(&Bs[0][0]) + (unsigned)(srow * LDT + shalf) * 2u;
  const unsigned short* gA = A  + (size_t)(rowB0 + srow) * K + shalf;
  const unsigned short* gB = Bt + (size_t)(colB0 + srow) * K + shalf;

  auto issue = [&](int kt, int buf) {
    const unsigned short* pa = gA + (size_t)kt * 32;
    const unsigned short* pb = gB + (size_t)kt * 32;
    unsigned la = ldsA0 + (unsigned)buf * BUFSZ;
    unsigned lb = ldsB0 + (unsigned)buf * BUFSZ;
    async_b128(la, pa);
    async_b128_o16(la, pa);
    async_b128(lb, pb);
    async_b128_o16(lb, pb);
  };

  v8f acc[4][2] = {};
  const int nstages = K / 32;
  issue(0, 0);
  for (int st = 0; st < nstages; ++st) {
    const int cur = st & 1;
    wait_async0();
    __syncthreads();
    if (st + 1 < nstages) issue(st + 1, cur ^ 1);

    const unsigned short* as = &As[cur][0];
    const unsigned short* bs = &Bs[cur][0];
    v16bf a[4], b[2];
#pragma unroll
    for (int i = 0; i < 4; ++i) {
      const unsigned short* p = as + (wm * 64 + i * 16 + m) * LDT + hi * 16;
      union { v16bf v; v8bf h[2]; } u;
      u.h[0] = *(const v8bf*)p;
      u.h[1] = *(const v8bf*)(p + 8);
      a[i] = u.v;
    }
#pragma unroll
    for (int j = 0; j < 2; ++j) {
      const unsigned short* p = bs + (wn * 32 + j * 16 + m) * LDT + hi * 16;
      union { v16bf v; v8bf h[2]; } u;
      u.h[0] = *(const v8bf*)p;
      u.h[1] = *(const v8bf*)(p + 8);
      b[j] = u.v;
    }
#pragma unroll
    for (int i = 0; i < 4; ++i)
#pragma unroll
      for (int j = 0; j < 2; ++j)
        acc[i][j] = __builtin_amdgcn_wmma_f32_16x16x32_bf16(
            false, a[i], false, b[j], (short)0, acc[i][j], false, false);
    __syncthreads();
  }

#pragma unroll
  for (int i = 0; i < 4; ++i)
#pragma unroll
    for (int j = 0; j < 2; ++j) {
      int col = colB0 + wn * 32 + j * 16 + m;
      float badd = bias ? bias[col] : 0.0f;
#pragma unroll
      for (int r = 0; r < 8; ++r) {
        int row = rowB0 + wm * 64 + i * 16 + r + 8 * hi;
        C[(size_t)row * Nn + col] = acc[i][j][r] + badd;
      }
    }
}

// ---------------------------------------------------------------- attention
// one workgroup per (b,h); 256 threads = 8 waves
__global__ __launch_bounds__(256) void attn_kernel(const float* __restrict__ qkv,
                                                   unsigned short* __restrict__ attn) {
  const int b = blockIdx.x >> 4;
  const int h = blockIdx.x & 15;
  const int hoff = h * DHEAD;
  const float* base = qkv + (size_t)b * NSEQ * QKV3;
  const int tid = threadIdx.x, lane = tid & 31, wave = tid >> 5;
  const int m = lane & 15, hi = lane >> 4;

  __shared__ float s_part[8][64][2];                 // per-wave (max,sum) per d
  __shared__ float s_kmax[64], s_kinv[64];
  __shared__ __align__(32) unsigned short Ks[64][80];    // ksm[d][n_local] bf16
  __shared__ __align__(32) unsigned short Vs[64][80];    // v[e][n_local]   bf16
  __shared__ __align__(32) unsigned short CtxT[64][80];  // ctx^T [e][d]    bf16
  __shared__ __align__(32) unsigned short Qs[64][80];    // qsm[n_local][d] bf16

  // ---- phase 1: k row stats (softmax over n per feature d), online
  {
    const int d0 = lane * 2;
    float m0 = -1e30f, s0 = 0.f, m1 = -1e30f, s1 = 0.f;
    for (int n = wave; n < NSEQ; n += 8) {
      const float* p = base + (size_t)n * QKV3 + HIDDEN + hoff + d0;
      float x0 = p[0], x1 = p[1];
      float nm0 = fmaxf(m0, x0); s0 = s0 * __expf(m0 - nm0) + __expf(x0 - nm0); m0 = nm0;
      float nm1 = fmaxf(m1, x1); s1 = s1 * __expf(m1 - nm1) + __expf(x1 - nm1); m1 = nm1;
    }
    s_part[wave][d0][0] = m0;   s_part[wave][d0][1] = s0;
    s_part[wave][d0+1][0] = m1; s_part[wave][d0+1][1] = s1;
  }
  __syncthreads();
  if (tid < 64) {
    float M = -1e30f, S = 0.f;
#pragma unroll
    for (int w = 0; w < 8; ++w) {
      float mw = s_part[w][tid][0], sw = s_part[w][tid][1];
      float nM = fmaxf(M, mw);
      S = S * __expf(M - nM) + sw * __expf(mw - nM);
      M = nM;
    }
    s_kmax[tid] = M;
    s_kinv[tid] = 1.0f / S;
  }
  __syncthreads();

  // fragment position ownership: wave owns (fm, fe0) and (fm, fe1)
  const int fm  = wave >> 1;
  const int fe0 = (2 * wave) & 3;
  const int fe1 = fe0 + 1;

  // ---- phase 2: context[d][e] = sum_n ksm[d,n] * v[e,n]   (WMMA over n)
  v8f cacc[2] = {};
  for (int chunk = 0; chunk < NSEQ / 64; ++chunk) {
    const int n0 = chunk * 64;
    {   // stage ksm and v tiles (coalesced channel reads)
      const int d  = tid & 63;
      const int nb = tid >> 6;
      const float kmaxd = s_kmax[d];
      const float kinvd = s_kinv[d];
#pragma unroll 4
      for (int pass = 0; pass < 16; ++pass) {
        const int nl = pass * 4 + nb;
        const float* rowp = base + (size_t)(n0 + nl) * QKV3;
        float kv = rowp[HIDDEN + hoff + d];
        float vv = rowp[2 * HIDDEN + hoff + d];
        Ks[d][nl] = f2bf(__expf(kv - kmaxd) * kinvd);
        Vs[d][nl] = f2bf(vv);
      }
    }
    __syncthreads();
#pragma unroll
    for (int ks = 0; ks < 2; ++ks) {
      const int k0 = ks * 32;
      union { v16bf v; v8bf hh[2]; } ua;
      const unsigned short* pr = &Ks[fm * 16 + m][0];
      ua.hh[0] = *(const v8bf*)(pr + k0 + hi * 16);
      ua.hh[1] = *(const v8bf*)(pr + k0 + hi * 16 + 8);
      v16bf b0 = *(const v16bf*)&Vs[fe0 * 16 + m][k0 + hi * 16];
      v16bf b1 = *(const v16bf*)&Vs[fe1 * 16 + m][k0 + hi * 16];
      cacc[0] = __builtin_amdgcn_wmma_f32_16x16x32_bf16(false, ua.v, false, b0, (short)0, cacc[0], false, false);
      cacc[1] = __builtin_amdgcn_wmma_f32_16x16x32_bf16(false, ua.v, false, b1, (short)0, cacc[1], false, false);
    }
    __syncthreads();
  }
  // write ctx^T (bf16) to LDS for phase 3
#pragma unroll
  for (int j = 0; j < 2; ++j) {
    const int fe = j ? fe1 : fe0;
#pragma unroll
    for (int r = 0; r < 8; ++r) {
      int d = fm * 16 + r + 8 * hi;
      int e = fe * 16 + m;
      CtxT[e][d] = f2bf(cacc[j][r]);
    }
  }
  __syncthreads();

  // ---- phase 3: out[n,e] = sum_d qsm[d,n]*SCALE * ctx[d,e]   (WMMA, K=d)
  for (int chunk = 0; chunk < NSEQ / 64; ++chunk) {
    const int n0 = chunk * 64;
    // stage q softmax rows: wave handles 8 n-rows, lanes span d
#pragma unroll
    for (int i = 0; i < 8; ++i) {
      const int nl = wave * 8 + i;
      const float* p = base + (size_t)(n0 + nl) * QKV3 + hoff + lane * 2;
      float x0 = p[0], x1 = p[1];
      float mx = fmaxf(x0, x1);
#pragma unroll
      for (int off = 16; off; off >>= 1) mx = fmaxf(mx, __shfl_xor(mx, off, 32));
      float e0 = __expf(x0 - mx), e1 = __expf(x1 - mx);
      float s = e0 + e1;
#pragma unroll
      for (int off = 16; off; off >>= 1) s += __shfl_xor(s, off, 32);
      float inv = ATT_SCALE / s;
      Qs[nl][lane * 2]     = f2bf(e0 * inv);
      Qs[nl][lane * 2 + 1] = f2bf(e1 * inv);
    }
    __syncthreads();

    v8f oacc[2] = {};
#pragma unroll
    for (int ks = 0; ks < 2; ++ks) {
      const int k0 = ks * 32;
      union { v16bf v; v8bf hh[2]; } ua;
      const unsigned short* pr = &Qs[fm * 16 + m][0];
      ua.hh[0] = *(const v8bf*)(pr + k0 + hi * 16);
      ua.hh[1] = *(const v8bf*)(pr + k0 + hi * 16 + 8);
      v16bf b0 = *(const v16bf*)&CtxT[fe0 * 16 + m][k0 + hi * 16];
      v16bf b1 = *(const v16bf*)&CtxT[fe1 * 16 + m][k0 + hi * 16];
      oacc[0] = __builtin_amdgcn_wmma_f32_16x16x32_bf16(false, ua.v, false, b0, (short)0, oacc[0], false, false);
      oacc[1] = __builtin_amdgcn_wmma_f32_16x16x32_bf16(false, ua.v, false, b1, (short)0, oacc[1], false, false);
    }
#pragma unroll
    for (int j = 0; j < 2; ++j) {
      const int fe = j ? fe1 : fe0;
#pragma unroll
      for (int r = 0; r < 8; ++r) {
        int n = n0 + fm * 16 + r + 8 * hi;
        int e = fe * 16 + m;
        attn[(size_t)(b * NSEQ + n) * HIDDEN + hoff + e] = f2bf(oacc[j][r]);
      }
    }
    __syncthreads();
  }
}

// ---------------------------------------------------------------- launch
extern "C" void kernel_launch(void* const* d_in, const int* in_sizes, int n_in,
                              void* d_out, int out_size, void* d_ws, size_t ws_size,
                              hipStream_t stream) {
  (void)in_sizes; (void)n_in; (void)out_size; (void)ws_size;
  const float* x     = (const float*)d_in[0];   // [4,4096,1024]
  const float* w_qkv = (const float*)d_in[1];   // [1024,3072]
  const float* w_out = (const float*)d_in[2];   // [1024,1024]
  const float* b_out = (const float*)d_in[3];   // [1024]
  float* out = (float*)d_out;                   // [4,4096,1024]

  const int M = 4 * NSEQ;                       // 16384 rows
  char* ws = (char*)d_ws;
  size_t off = 0;
  unsigned short* x_bf   = (unsigned short*)(ws + off); off += (size_t)M * HIDDEN * 2;       // 32MB
  unsigned short* wqkvT  = (unsigned short*)(ws + off); off += (size_t)QKV3 * HIDDEN * 2;    // 6MB
  unsigned short* woutT  = (unsigned short*)(ws + off); off += (size_t)HIDDEN * HIDDEN * 2;  // 2MB
  float*          qkv    = (float*)(ws + off);          off += (size_t)M * QKV3 * 4;         // 192MB
  unsigned short* attn   = (unsigned short*)(ws + off);                                      // 32MB

  cvt_bf16_kernel<<<(M * HIDDEN) / 256, 256, 0, stream>>>(x, x_bf, M * HIDDEN);
  cvt_bf16_T_kernel<<<(HIDDEN * QKV3) / 256, 256, 0, stream>>>(w_qkv, wqkvT, HIDDEN, QKV3);
  cvt_bf16_T_kernel<<<(HIDDEN * HIDDEN) / 256, 256, 0, stream>>>(w_out, woutT, HIDDEN, HIDDEN);

  gemm_bf16_kernel<<<dim3(QKV3 / 128, M / 128), 256, 0, stream>>>(
      x_bf, wqkvT, qkv, nullptr, M, QKV3, HIDDEN);

  attn_kernel<<<4 * HEADS, 256, 0, stream>>>(qkv, attn);

  gemm_bf16_kernel<<<dim3(HIDDEN / 128, M / 128), 256, 0, stream>>>(
      attn, woutT, out, b_out, M, HIDDEN, HIDDEN);
}
